// ColumnMixedPrecisionLinear_1984274891497
// MI455X (gfx1250) — compile-verified, matrix operand order checked by reading, not run on hardware
//
#include <hip/hip_runtime.h>
#include <stdint.h>

typedef __attribute__((ext_vector_type(16))) _Float16 v16h;
typedef __attribute__((ext_vector_type(8)))  _Float16 v8h;
typedef __attribute__((ext_vector_type(8)))  float    v8f;
typedef __attribute__((ext_vector_type(4)))  unsigned int u32x4;
typedef __attribute__((ext_vector_type(8)))  int          i32x8;
typedef __attribute__((ext_vector_type(4)))  int          i32x4;

#define D_IN_SLICE 1024
#define K_TOTAL    4096
#define D_OUT      4096
#define M_TOTAL    8192

// main GEMM tiling
#define BM  128
#define BN  128
#define BK  64
#define LDT 72            // LDS row stride in halves: 64 data + 8 pad (144B rows, 16B aligned)

#if __has_builtin(__builtin_amdgcn_tensor_load_to_lds)
#define USE_TDM 1
#else
#define USE_TDM 0
#endif

// ---------------------------------------------------------------------------
// Prep pass 1: x fp32 -> f16  (bandwidth bound: 128MB rd + 64MB wr ~ 8us)
// ---------------------------------------------------------------------------
__global__ __launch_bounds__(256)
void convert_x_f16(const float* __restrict__ x, _Float16* __restrict__ xh)
{
    const size_t i = ((size_t)blockIdx.x * 256 + threadIdx.x) * 8;
    const float4 a = *reinterpret_cast<const float4*>(x + i);
    const float4 b = *reinterpret_cast<const float4*>(x + i + 4);
    v8h h;
    h[0] = (_Float16)a.x; h[1] = (_Float16)a.y; h[2] = (_Float16)a.z; h[3] = (_Float16)a.w;
    h[4] = (_Float16)b.x; h[5] = (_Float16)b.y; h[6] = (_Float16)b.z; h[7] = (_Float16)b.w;
    *reinterpret_cast<v8h*>(xh + i) = h;
}

// ---------------------------------------------------------------------------
// Prep pass 2: wh[o][sl*1024 + k] = f16(wq_sl[o][k] * s_sl[o])
// Folds the per-slice per-channel scale into the weights once, so the main
// loop is a single plain f16 GEMM with one accumulator set.
// ---------------------------------------------------------------------------
__global__ __launch_bounds__(256)
void dequant_w_f16(const int8_t* __restrict__ wq, const float* __restrict__ s,
                   _Float16* __restrict__ wh, int sl)
{
    const int idx = blockIdx.x * 256 + threadIdx.x;   // 4096 rows * 64 chunks
    const int o   = idx >> 6;
    const int c16 = (idx & 63) << 4;
    union { int4 v; int8_t b[16]; } u;
    u.v = *reinterpret_cast<const int4*>(wq + (size_t)o * D_IN_SLICE + c16);
    const float sc = s[o];
    _Float16* dst = wh + (size_t)o * K_TOTAL + sl * D_IN_SLICE + c16;
    v8h h0, h1;
    #pragma unroll
    for (int j = 0; j < 8; ++j) h0[j] = (_Float16)((float)u.b[j] * sc);
    #pragma unroll
    for (int j = 0; j < 8; ++j) h1[j] = (_Float16)((float)u.b[8 + j] * sc);
    *reinterpret_cast<v8h*>(dst)     = h0;
    *reinterpret_cast<v8h*>(dst + 8) = h1;
}

// ---------------------------------------------------------------------------
// TDM descriptor: 2D f16 tile (tile_cols x tile_rows) -> LDS, with LDS row
// padding (32 dwords data + 4 dwords pad -> 144B effective stride == LDT).
// D# packing per ISA 8.3/8.4: count=1, type=2, data_size=1 (2 bytes).
// ---------------------------------------------------------------------------
#if USE_TDM
__device__ __forceinline__ void tdm_load_tile_f16(uint32_t lds_byte,
                                                  const _Float16* gtile,
                                                  uint32_t tensor_rows,
                                                  uint32_t row_len_elems,
                                                  uint32_t tile_cols,
                                                  uint32_t tile_rows)
{
    const uint64_t ga = (uint64_t)(uintptr_t)gtile;
    u32x4 g0;
    g0.x = 1u;                                                   // count=1, user mode
    g0.y = lds_byte;                                             // lds_addr
    g0.z = (uint32_t)ga;                                         // global_addr[31:0]
    g0.w = (uint32_t)((ga >> 32) & 0x01ffffffu) | (2u << 30);    // [56:32] | type=2
    // pad_enable, pad_interval code 4 (32 dwords = one 128B row), pad_amount code 3 (4 dwords)
    const uint32_t flags = (1u << 16) | (1u << 20) | (4u << 22) | (3u << 25);
    i32x8 g1;
    g1[0] = (int)flags;
    g1[1] = (int)((row_len_elems & 0xffffu) << 16);                       // tensor_dim0 lo
    g1[2] = (int)((row_len_elems >> 16) | ((tensor_rows & 0xffffu) << 16)); // dim0 hi | dim1 lo
    g1[3] = (int)((tensor_rows >> 16) | (tile_cols << 16));               // dim1 hi | tile_dim0
    g1[4] = (int)tile_rows;                                               // tile_dim1 (tile_dim2=0)
    g1[5] = (int)row_len_elems;                                           // tensor_dim0_stride lo
    g1[6] = 0;
    g1[7] = 0;
    const i32x4 z4 = {0, 0, 0, 0};
#if __has_include(<hip/amd_detail/amd_gfx1250_TDM.h>)
    const i32x8 z8 = {0, 0, 0, 0, 0, 0, 0, 0};
    __builtin_amdgcn_tensor_load_to_lds(g0, g1, z4, z4, z8, 0);   // clang-23 form
#else
    __builtin_amdgcn_tensor_load_to_lds(g0, g1, z4, z4, 0);       // ROCm 7.2 form
#endif
}
#endif

// ---------------------------------------------------------------------------
// Main GEMM: out[M,N] = xh[M,K] * wh[N,K]^T + bias   (f16 in, f32 accum)
// 8 wave32 per WG; wave owns 32x64 (2x4 WMMA frags); BK=64 double buffered.
// ---------------------------------------------------------------------------
__global__ __launch_bounds__(256)
void gemm_f16_tdm(const _Float16* __restrict__ xh,
                  const _Float16* __restrict__ wh,
                  const float* __restrict__ bias,
                  float* __restrict__ out)
{
    __shared__ _Float16 As[2][BM * LDT];   // 2 x 18432 B
    __shared__ _Float16 Bs[2][BN * LDT];   // 2 x 18432 B

    const int tid  = threadIdx.x;
    const int lane = tid & 31;
    const int wave = tid >> 5;       // 0..7
    const int mw   = wave & 3;       // 32-row band of 128
    const int nw   = wave >> 2;      // 64-col band of 128

    const int m0 = blockIdx.x * BM;
    const int n0 = blockIdx.y * BN;

    // WMMA wave32 fragment addressing (ISA 7.12.2)
    const int frow = lane & 15;
    const int khi  = (lane >> 4) << 3;   // 0 or 8 halves
    const int mhi  = (lane >> 4) << 3;   // C/D: lanes>=16 hold M = v+8

    v8f acc[2][4];
    #pragma unroll
    for (int i = 0; i < 2; ++i)
        #pragma unroll
        for (int j = 0; j < 4; ++j)
            acc[i][j] = (v8f){0.f,0.f,0.f,0.f,0.f,0.f,0.f,0.f};

    auto stage = [&](int buf, int kb) {
#if USE_TDM
        if (wave == 0) {
            tdm_load_tile_f16((uint32_t)(uintptr_t)&As[buf][0],
                              xh + (size_t)m0 * K_TOTAL + kb,
                              M_TOTAL, K_TOTAL, BK, BM);
            tdm_load_tile_f16((uint32_t)(uintptr_t)&Bs[buf][0],
                              wh + (size_t)n0 * K_TOTAL + kb,
                              D_OUT, K_TOTAL, BK, BN);
        }
#else
        #pragma unroll
        for (int it = 0; it < 4; ++it) {            // A: 128 rows x 8 16B chunks
            const int i = tid + it * 256;
            const int r = i >> 3, c8 = (i & 7) << 3;
            *reinterpret_cast<v8h*>(&As[buf][r * LDT + c8]) =
                *reinterpret_cast<const v8h*>(xh + (size_t)(m0 + r) * K_TOTAL + kb + c8);
        }
        #pragma unroll
        for (int it = 0; it < 4; ++it) {            // B: 128 rows x 8 16B chunks
            const int i = tid + it * 256;
            const int r = i >> 3, c8 = (i & 7) << 3;
            *reinterpret_cast<v8h*>(&Bs[buf][r * LDT + c8]) =
                *reinterpret_cast<const v8h*>(wh + (size_t)(n0 + r) * K_TOTAL + kb + c8);
        }
#endif
    };
    auto wait_stage = [&]() {
#if USE_TDM
        if (wave == 0) __builtin_amdgcn_s_wait_tensorcnt(0);
#endif
    };

    stage(0, 0);
    wait_stage();
    __syncthreads();

    for (int kb = 0; kb < K_TOTAL; kb += BK) {
        const int cur = (kb >> 6) & 1;
        if (kb + BK < K_TOTAL) stage(cur ^ 1, kb + BK);   // DMA next tile during compute

        const _Float16* __restrict__ Ab = &As[cur][0];
        const _Float16* __restrict__ Bb = &Bs[cur][0];
        #pragma unroll
        for (int kk = 0; kk < BK; kk += 32) {
            v16h afr[2], bfr[4];
            #pragma unroll
            for (int mt = 0; mt < 2; ++mt) {
                const _Float16* ap = Ab + (mw * 32 + mt * 16 + frow) * LDT + kk + khi;
                const v8h lo = *reinterpret_cast<const v8h*>(ap);
                const v8h hi = *reinterpret_cast<const v8h*>(ap + 16);
                afr[mt] = __builtin_shufflevector(lo, hi,
                    0,1,2,3,4,5,6,7,8,9,10,11,12,13,14,15);
            }
            #pragma unroll
            for (int nt = 0; nt < 4; ++nt) {
                const _Float16* bp = Bb + (nw * 64 + nt * 16 + frow) * LDT + kk + khi;
                const v8h lo = *reinterpret_cast<const v8h*>(bp);
                const v8h hi = *reinterpret_cast<const v8h*>(bp + 16);
                bfr[nt] = __builtin_shufflevector(lo, hi,
                    0,1,2,3,4,5,6,7,8,9,10,11,12,13,14,15);
            }
            #pragma unroll
            for (int mt = 0; mt < 2; ++mt)
                #pragma unroll
                for (int nt = 0; nt < 4; ++nt)
                    acc[mt][nt] = __builtin_amdgcn_wmma_f32_16x16x32_f16(
                        false, afr[mt], false, bfr[nt],
                        (short)0, acc[mt][nt], false, false);
        }

        wait_stage();        // next buffer's DMA done (overlapped with the WMMAs above)
        __syncthreads();
    }

    // epilogue: bias + fp32 store
    #pragma unroll
    for (int nt = 0; nt < 4; ++nt) {
        const int   n  = n0 + nw * 64 + nt * 16 + frow;
        const float bz = bias[n];
        #pragma unroll
        for (int mt = 0; mt < 2; ++mt) {
            const int mb = m0 + mw * 32 + mt * 16 + mhi;
            #pragma unroll
            for (int v = 0; v < 8; ++v)
                out[(size_t)(mb + v) * D_OUT + n] = acc[mt][nt][v] + bz;
        }
    }
}

// ---------------------------------------------------------------------------
extern "C" void kernel_launch(void* const* d_in, const int* in_sizes, int n_in,
                              void* d_out, int out_size, void* d_ws, size_t ws_size,
                              hipStream_t stream) {
    (void)in_sizes; (void)n_in; (void)out_size; (void)ws_size;
    const float*  x    = (const float*)d_in[0];
    const int8_t* wq0  = (const int8_t*)d_in[1];
    const float*  s0   = (const float*)d_in[2];
    const int8_t* wq1  = (const int8_t*)d_in[3];
    const float*  s1   = (const float*)d_in[4];
    const int8_t* wq2  = (const int8_t*)d_in[5];
    const float*  s2   = (const float*)d_in[6];
    const int8_t* wq3  = (const int8_t*)d_in[7];
    const float*  s3   = (const float*)d_in[8];
    const float*  bias = (const float*)d_in[9];
    float* out = (float*)d_out;

    // workspace: [0,64MB) x as f16, [64MB,96MB) pre-scaled weights as f16
    _Float16* xh = (_Float16*)d_ws;
    _Float16* wh = (_Float16*)((char*)d_ws + (size_t)M_TOTAL * K_TOTAL * sizeof(_Float16));

    convert_x_f16<<<dim3((M_TOTAL * (size_t)K_TOTAL) / 8 / 256), dim3(256), 0, stream>>>(x, xh);
    dequant_w_f16<<<dim3(1024), dim3(256), 0, stream>>>(wq0, s0, wh, 0);
    dequant_w_f16<<<dim3(1024), dim3(256), 0, stream>>>(wq1, s1, wh, 1);
    dequant_w_f16<<<dim3(1024), dim3(256), 0, stream>>>(wq2, s2, wh, 2);
    dequant_w_f16<<<dim3(1024), dim3(256), 0, stream>>>(wq3, s3, wh, 3);

    gemm_f16_tdm<<<dim3(M_TOTAL / BM, D_OUT / BN), dim3(256), 0, stream>>>(xh, wh, bias, out);
}
